// GNN_2448131359246
// MI455X (gfx1250) — matrix operand order repaired and checked by reading, not compiled, not run on hardware
//
#include <hip/hip_runtime.h>
#include <stdint.h>

#define IN_DIM 256
#define HID    300
#define HIDP   304                 // padded hidden dim (19 tiles of 16; 76 float4 per row)
#define NT     (HIDP / 16)         // 19 N-tiles

typedef __attribute__((ext_vector_type(16))) __bf16       v16bf;
typedef __attribute__((ext_vector_type(8)))  float        v8f;
typedef __attribute__((ext_vector_type(4)))  unsigned int v4u;
typedef __attribute__((ext_vector_type(8)))  unsigned int v8u;

union AFrag { v16bf bf; struct { v4u lo; v4u hi; } q; };
union BFrag { v16bf bf; v8u q; };

static __device__ __forceinline__ unsigned short f32_to_bf16(float f) {
  union { float f; unsigned u; } x; x.f = f;
  unsigned r = x.u + 0x7FFFu + ((x.u >> 16) & 1u);   // round-to-nearest-even
  return (unsigned short)(r >> 16);
}

static __device__ __forceinline__ void fma4(float4& a, const float4& v, float s) {
  a.x = fmaf(v.x, s, a.x); a.y = fmaf(v.y, s, a.y);
  a.z = fmaf(v.z, s, a.z); a.w = fmaf(v.w, s, a.w);
}

// ---- x (f32) -> bf16 -------------------------------------------------------
__global__ void k_cvt_x(const float* __restrict__ x, unsigned short* __restrict__ xb, int n) {
  int i = blockIdx.x * blockDim.x + threadIdx.x;
  if (i < n) xb[i] = f32_to_bf16(x[i]);
}

// ---- W1 [256,300] -> W1^T bf16 [304,256], zero-padded rows 300..303 --------
__global__ void k_cvt_w(const float* __restrict__ W1, unsigned short* __restrict__ w1t) {
  int i = blockIdx.x * blockDim.x + threadIdx.x;
  if (i >= HIDP * IN_DIM) return;
  int n = i / IN_DIM, k = i - n * IN_DIM;
  w1t[i] = (n < HID) ? f32_to_bf16(W1[k * HID + n]) : (unsigned short)0;
}

// ---- in-degree histogram (int atomics) -------------------------------------
__global__ void k_deg_init(int* __restrict__ degi, int N) {
  int i = blockIdx.x * blockDim.x + threadIdx.x;
  if (i < N) degi[i] = 0;
}

__global__ void k_deg(const int* __restrict__ dst, int* __restrict__ degi, int E) {
  int i = blockIdx.x * blockDim.x + threadIdx.x;
  if (i < E) atomicAdd(&degi[dst[i]], 1);
}

__global__ void k_dis(const int* __restrict__ degi, float* __restrict__ dis, int N) {
  int i = blockIdx.x * blockDim.x + threadIdx.x;
  if (i < N) dis[i] = rsqrtf((float)degi[i] + 1.0f);   // +1 = self loop
}

// ---- exclusive scan of degi -> rowptr[0..N] (single workgroup, LDS scan) ---
__global__ void k_scan(const int* __restrict__ degi, int* __restrict__ rowptr, int N) {
  __shared__ int sm[1024];
  __shared__ int running;
  if (threadIdx.x == 0) running = 0;
  __syncthreads();
  for (int base = 0; base < N; base += 1024) {
    int i = base + threadIdx.x;
    int v = (i < N) ? degi[i] : 0;
    sm[threadIdx.x] = v;
    __syncthreads();
    for (int off = 1; off < 1024; off <<= 1) {     // Hillis-Steele inclusive scan
      int t = (threadIdx.x >= (unsigned)off) ? sm[threadIdx.x - off] : 0;
      __syncthreads();
      sm[threadIdx.x] += t;
      __syncthreads();
    }
    if (i < N) rowptr[i] = running + sm[threadIdx.x] - v;   // exclusive
    __syncthreads();
    if (threadIdx.x == 0) running += sm[1023];
    __syncthreads();
  }
  if (threadIdx.x == 0) rowptr[N] = running;       // == E
}

__global__ void k_cursor(const int* __restrict__ rowptr, int* __restrict__ cursor, int N) {
  int i = blockIdx.x * blockDim.x + threadIdx.x;
  if (i < N) cursor[i] = rowptr[i];
}

// ---- CSR fill: one int atomic per edge -------------------------------------
__global__ void k_fill(const int* __restrict__ src, const int* __restrict__ dst,
                       int* __restrict__ cursor, int* __restrict__ csrSrc, int E) {
  int e = blockIdx.x * blockDim.x + threadIdx.x;
  if (e < E) {
    int p = atomicAdd(&cursor[dst[e]], 1);
    csrSrc[p] = src[e];
  }
}

// ---- h = x @ W1 via v_wmma_f32_16x16x32_bf16 -------------------------------
// One 16x16 output tile per wave; K=256 in 8 steps of 32.
__global__ void __launch_bounds__(128) k_gemm(const unsigned short* __restrict__ xb,
                                              const unsigned short* __restrict__ w1t,
                                              float* __restrict__ h, int N) {
  int wave = threadIdx.x >> 5;
  int lane = threadIdx.x & 31;
  int tile = blockIdx.x * 4 + wave;
  int mtiles = N >> 4;                            // N % 16 == 0 (50000 = 3125*16)
  if (tile >= mtiles * NT) return;                // wave-uniform: EXEC stays all-1s
  int mt = tile / NT, nt = tile - mt * NT;

  int hi = lane >> 4;                             // lane half (0/1)
  int l16 = lane & 15;
  int arow = mt * 16 + l16;                       // A row for this lane
  int ncol = nt * 16 + l16;                       // B column for this lane
  // A 16x32 bf16 fragment: lane half 0 -> K {0..15}, half 1 -> K {16..31}
  const unsigned short* aBase = xb  + (size_t)arow * IN_DIM + hi * 16;
  // B 32x16 bf16 fragment (W1^T row-major => 16 contiguous K per lane)
  const unsigned short* bBase = w1t + (size_t)ncol * IN_DIM + hi * 16;

  v8f acc = {};
#pragma unroll
  for (int k0 = 0; k0 < IN_DIM; k0 += 32) {
    __builtin_prefetch(aBase + k0 + 128, 0, 0);   // global_prefetch_b8 (speculative)
    AFrag a; BFrag b;
    a.q.lo = *(const v4u*)(aBase + k0);           // 16B: K=k0+hi*16 .. +7
    a.q.hi = *(const v4u*)(aBase + k0 + 8);       // 16B: K=k0+hi*16+8 .. +15
    b.q    = *(const v8u*)(bBase + k0);           // 32B: 16 contiguous K
    acc = __builtin_amdgcn_wmma_f32_16x16x32_bf16(
        /*neg_a=*/false, a.bf, /*neg_b=*/false, b.bf,
        /*c_mod=*/(short)0, acc, /*reuse_a=*/false, /*reuse_b=*/false);
  }
  // C layout: VGPR r -> M = mt*16 + hi*8 + r, N = ncol
  int rbase = mt * 16 + hi * 8;
#pragma unroll
  for (int r = 0; r < 8; ++r)
    h[(size_t)(rbase + r) * HIDP + ncol] = acc[r];
}

// ---- agg = b1 + self-loop + gather over CSR (no float atomics) -------------
// One wave32 per node; each lane keeps 3 float4 accumulators (covers 304 cols).
__global__ void __launch_bounds__(256) k_aggregate(const int* __restrict__ rowptr,
                                                   const int* __restrict__ csrSrc,
                                                   const float* __restrict__ h,
                                                   const float* __restrict__ dis,
                                                   const float* __restrict__ b1,
                                                   float* __restrict__ agg, int N) {
  int wid  = (blockIdx.x * blockDim.x + threadIdx.x) >> 5;
  int lane = threadIdx.x & 31;
  if (wid >= N) return;                           // wave-uniform
  float dn = dis[wid];
  bool t2 = (lane < 12);                          // tail chunk: f4 = 64..75 (<= 303, pads are 0)
  float4 a0 = {0,0,0,0}, a1 = {0,0,0,0}, a2 = {0,0,0,0};

  int beg = rowptr[wid], end = rowptr[wid + 1];
  for (int j = beg; j < end; ++j) {
    int s = csrSrc[j];
    float nrm = dis[s] * dn;
    const float4* hp = (const float4*)(h + (size_t)s * HIDP);
    float4 v0 = hp[lane];
    float4 v1 = hp[lane + 32];
    fma4(a0, v0, nrm);
    fma4(a1, v1, nrm);
    if (t2) { float4 v2 = hp[lane + 64]; fma4(a2, v2, nrm); }
  }

  // self loop: h[wid] * dn^2
  float sn = dn * dn;
  const float4* hs = (const float4*)(h + (size_t)wid * HIDP);
  fma4(a0, hs[lane], sn);
  fma4(a1, hs[lane + 32], sn);
  if (t2) fma4(a2, hs[lane + 64], sn);

  // bias (b1 has exactly 300 floats = 75 float4; lane 11 tail covers pads only)
  const float4* b4 = (const float4*)b1;
  float4 c0 = b4[lane], c1 = b4[lane + 32];
  a0.x += c0.x; a0.y += c0.y; a0.z += c0.z; a0.w += c0.w;
  a1.x += c1.x; a1.y += c1.y; a1.z += c1.z; a1.w += c1.w;
  if (lane < 11) { float4 c2 = b4[lane + 64];
    a2.x += c2.x; a2.y += c2.y; a2.z += c2.z; a2.w += c2.w; }

  float4* ap = (float4*)(agg + (size_t)wid * HIDP);
  ap[lane] = a0;
  ap[lane + 32] = a1;
  if (t2) ap[lane + 64] = a2;
}

// ---- out = relu(agg) @ Wf + bf : one wave32 per node -----------------------
__global__ void __launch_bounds__(256) k_final(const float* __restrict__ agg,
                                               const float* __restrict__ Wf,
                                               const float* __restrict__ bfv,
                                               float* __restrict__ out, int N) {
  int wid  = (blockIdx.x * blockDim.x + threadIdx.x) >> 5;
  int lane = threadIdx.x & 31;
  if (wid >= N) return;
  const float* a = agg + (size_t)wid * HIDP;
  float s0 = 0.0f, s1 = 0.0f;
  for (int f = lane; f < HID; f += 32) {
    float v = a[f]; v = v > 0.0f ? v : 0.0f;
    s0 += v * Wf[f * 2 + 0];
    s1 += v * Wf[f * 2 + 1];
  }
#pragma unroll
  for (int off = 16; off > 0; off >>= 1) {
    s0 += __shfl_xor(s0, off, 32);
    s1 += __shfl_xor(s1, off, 32);
  }
  if (lane == 0) { out[wid * 2 + 0] = s0 + bfv[0]; out[wid * 2 + 1] = s1 + bfv[1]; }
}

extern "C" void kernel_launch(void* const* d_in, const int* in_sizes, int n_in,
                              void* d_out, int out_size, void* d_ws, size_t ws_size,
                              hipStream_t stream) {
  const float* x   = (const float*)d_in[0];
  const int*   eix = (const int*)d_in[1];       // edge_index flat [2,E], int32 per harness ABI
  const float* W1  = (const float*)d_in[2];
  const float* b1  = (const float*)d_in[3];
  const float* Wf  = (const float*)d_in[4];
  const float* bfv = (const float*)d_in[5];
  float* out = (float*)d_out;

  int N = in_sizes[0] / IN_DIM;                  // 50000
  int E = in_sizes[1] / 2;                       // 800000
  const int* src = eix;
  const int* dst = eix + E;

  // Carve workspace (d_ws is 256B-aligned from hipMalloc).
  char* ws = (char*)d_ws; size_t off = 0;
  auto alloc = [&](size_t bytes) -> char* {
    char* p = ws + off; off = (off + bytes + 255) & ~(size_t)255; return p;
  };
  unsigned short* xb     = (unsigned short*)alloc((size_t)N * IN_DIM * 2);   // 25.6 MB
  unsigned short* w1t    = (unsigned short*)alloc((size_t)HIDP * IN_DIM * 2);
  float* h      = (float*)alloc((size_t)N * HIDP * 4);                       // 60.8 MB
  float* agg    = (float*)alloc((size_t)N * HIDP * 4);                       // 60.8 MB
  int*   degi   = (int*)alloc((size_t)N * 4);
  int*   rowptr = (int*)alloc(((size_t)N + 1) * 4);
  int*   cursor = (int*)alloc((size_t)N * 4);
  int*   csrSrc = (int*)alloc((size_t)E * 4);                                // 3.2 MB
  float* dis    = (float*)alloc((size_t)N * 4);
  (void)ws_size; (void)n_in; (void)out_size;

  const int T = 256;
  k_cvt_x   <<<(N * IN_DIM + T - 1) / T, T, 0, stream>>>(x, xb, N * IN_DIM);
  k_cvt_w   <<<(HIDP * IN_DIM + T - 1) / T, T, 0, stream>>>(W1, w1t);
  k_deg_init<<<(N + T - 1) / T, T, 0, stream>>>(degi, N);
  k_deg     <<<(E + T - 1) / T, T, 0, stream>>>(dst, degi, E);
  k_dis     <<<(N + T - 1) / T, T, 0, stream>>>(degi, dis, N);
  k_scan    <<<1, 1024, 0, stream>>>(degi, rowptr, N);
  k_cursor  <<<(N + T - 1) / T, T, 0, stream>>>(rowptr, cursor, N);
  k_fill    <<<(E + T - 1) / T, T, 0, stream>>>(src, dst, cursor, csrSrc, E);

  int tiles = (N / 16) * NT;                     // 3125 * 19
  k_gemm    <<<(tiles + 3) / 4, 128, 0, stream>>>(xb, w1t, h, N);

  k_aggregate<<<(N + 7) / 8, T, 0, stream>>>(rowptr, csrSrc, h, dis, b1, agg, N);
  k_final    <<<(N + 7) / 8, T, 0, stream>>>(agg, Wf, bfv, out, N);
}
